// MobileViTAttention_46866683134140
// MI455X (gfx1250) — matrix operand order repaired and checked by reading, not compiled, not run on hardware
//
#include <hip/hip_runtime.h>

// ---------------- CDNA5 WMMA / TDM types ----------------
typedef __attribute__((ext_vector_type(16))) __bf16 v16bf;
typedef __attribute__((ext_vector_type(8)))  __bf16 v8bf;
typedef __attribute__((ext_vector_type(4)))  __bf16 v4bf;
typedef __attribute__((ext_vector_type(2)))  __bf16 v2bf;
typedef __attribute__((ext_vector_type(8)))  float  v8f;
typedef __attribute__((ext_vector_type(4)))  unsigned int v4u;
typedef __attribute__((ext_vector_type(8)))  int v8i;
typedef __attribute__((ext_vector_type(4)))  int v4i;

#define BATCH 16
#define HW    28
#define NPIX  784          // 28*28
#define RWIN  49           // 7*7 windows
#define NH    8            // heads
#define HD    128          // head dim

static __device__ __forceinline__ v16bf cat16(v8bf lo, v8bf hi) {
    return __builtin_shufflevector(lo, hi, 0,1,2,3,4,5,6,7,8,9,10,11,12,13,14,15);
}

// Issue one TDM 2D tile load: 256 rows x 32 bf16 (64B/row), row stride K elems.
// Wave-level op (EXEC ignored), tracked by TENSORcnt.
static __device__ __forceinline__ void tdm_load_tile_A(const __bf16* src, unsigned lds, int K)
{
    unsigned long long ga = (unsigned long long)(uintptr_t)src;
    v4u g0;
    g0[0] = 1u;                                                  // count=1 (valid user D#)
    g0[1] = lds;                                                 // lds_addr
    g0[2] = (unsigned)(ga & 0xffffffffu);                        // global_addr[31:0]
    g0[3] = (unsigned)((ga >> 32) & 0x01ffffffu) | 0x80000000u;  // addr[56:32] | type=2
    v8i g1;
    g1[0] = 0x00010000;        // workgroup_mask=0, data_size=1 (2 bytes)
    g1[1] = (K & 0xffff) << 16;// tensor_dim0[15:0] = K (elems)
    g1[2] = (256 << 16);       // tensor_dim1[15:0] = 256 rows
    g1[3] = (32 << 16);        // tile_dim0 = 32 elems
    g1[4] = 256;               // tile_dim1 = 256, tile_dim2 = 0
    g1[5] = K;                 // tensor_dim0_stride[31:0] = K (elems)
    g1[6] = 0;
    g1[7] = 0;
    v4i z4 = {0, 0, 0, 0};
    v8i z8 = {0, 0, 0, 0, 0, 0, 0, 0};
    __builtin_amdgcn_tensor_load_to_lds(g0, g1, z4, z4, z8, 0);
}

// =====================================================================
// One-time f32 -> bf16 weight conversion (row-major preserved).
// =====================================================================
__global__ __launch_bounds__(256)
void f32_to_bf16_kernel(const float* __restrict__ in, __bf16* __restrict__ out, int n)
{
    int i = (blockIdx.x * 256 + threadIdx.x) * 8;
    if (i + 8 <= n) {
        float4 a = *(const float4*)(in + i);
        float4 b = *(const float4*)(in + i + 4);
        v8bf o;
        o[0] = (__bf16)a.x; o[1] = (__bf16)a.y; o[2] = (__bf16)a.z; o[3] = (__bf16)a.w;
        o[4] = (__bf16)b.x; o[5] = (__bf16)b.y; o[6] = (__bf16)b.z; o[7] = (__bf16)b.w;
        *(v8bf*)(out + i) = o;
    } else {
        for (int j = i; j < n; ++j) out[j] = (__bf16)in[j];
    }
}

// =====================================================================
// Batched GEMM: O[b][m][n] = sum_k W[m][k]*X[b][k][n] + bias[m] (+resid)
//   W pre-converted to bf16 row-major.
// Block: 256 threads (8 waves). Block tile 256(M) x 64(N), K step 32.
// Wave w: rows [32w, 32w+32), 4 N-subtiles -> 8 wmma / K-step.
// A tile (256x32 bf16, 16KB) staged by ONE tensor_load_to_lds (TDM) from
// wave 0; B tile staged b128-vectorized + transposed to [n][k] with packed
// bf16 ds_store_b32; bias slice staged by global_load_async_to_lds_b128.
// Double-buffered LDS; s_wait_tensorcnt + one barrier per K-step.
// =====================================================================
__global__ __launch_bounds__(256)
void wmma_gemm_kernel(const __bf16* __restrict__ Wbf,
                      const float* __restrict__ X,
                      const float* __restrict__ bias,
                      const float* __restrict__ resid,
                      float* __restrict__ O,
                      int M, int K, int N)
{
    __shared__ __bf16 As[2][256 * 32];   // [m][k] row-major, 16KB each
    __shared__ __bf16 Bs[2][64 * 32];    // [n][k], 4KB each
    __shared__ float  BiasLds[256];

    const int tid  = threadIdx.x;
    const int wave = tid >> 5;
    const int lane = tid & 31;
    const int half = lane >> 4;
    const int lm   = lane & 15;

    const int b     = blockIdx.z;
    const int mbase = blockIdx.y * 256;
    const int nbase = blockIdx.x * 64;
    const bool edge = (nbase + 64 > N);          // block-uniform

    const float* Xb = X + (size_t)b * K * N;

    // ---- bias slice via async-to-LDS (waves 0-1, full EXEC) ----
    if (tid < 64) {
        unsigned lds = (unsigned)(uintptr_t)&BiasLds[tid * 4];
        unsigned long long ga = (unsigned long long)(uintptr_t)(bias + mbase + tid * 4);
        asm volatile("global_load_async_to_lds_b128 %0, %1, off"
                     :: "v"(lds), "v"(ga) : "memory");
    }

    // ---- A staging: one TDM tile load from wave 0 ----
    auto stageA = [&](int buf, int k0) {
        if (wave == 0)
            tdm_load_tile_A(Wbf + (size_t)mbase * K + k0,
                            (unsigned)(uintptr_t)&As[buf][0], K);
    };

    // ---- B staging: thread t handles a 2(k) x 4(n) sub-block ----
    auto stageB = [&](int buf, int k0) {
        const int kb  = (tid & 15) * 2;       // 0..30
        const int nb  = (tid >> 4) * 4;       // 0..60
        const int gn0 = nbase + nb;
        const float* r0 = Xb + (size_t)(k0 + kb) * N + gn0;
        const float* r1 = r0 + N;
        float x0[4], x1[4];
        if (!edge) {
            float4 a = *(const float4*)r0;
            float4 c = *(const float4*)r1;
            x0[0]=a.x; x0[1]=a.y; x0[2]=a.z; x0[3]=a.w;
            x1[0]=c.x; x1[1]=c.y; x1[2]=c.z; x1[3]=c.w;
        } else {
            #pragma unroll
            for (int n = 0; n < 4; ++n) {
                bool ok = (gn0 + n) < N;
                x0[n] = ok ? r0[n] : 0.0f;
                x1[n] = ok ? r1[n] : 0.0f;
            }
        }
        if (k0 + 64 < K) __builtin_prefetch(r0 + (size_t)64 * N, 0, 0);
        #pragma unroll
        for (int n = 0; n < 4; ++n) {
            v2bf p; p[0] = (__bf16)x0[n]; p[1] = (__bf16)x1[n];
            *(v2bf*)(&Bs[buf][(nb + n) * 32 + kb]) = p;  // packed along k
        }
    };

    v8f acc[2][4] = {};

    stageA(0, 0);
    stageB(0, 0);
    asm volatile("s_wait_asynccnt 0x0" ::: "memory");   // bias slice
    if (wave == 0) __builtin_amdgcn_s_wait_tensorcnt(0);
    __syncthreads();

    int cur = 0;
    for (int k0 = 0; k0 < K; k0 += 32) {
        const int nxt = cur ^ 1;
        if (k0 + 32 < K) {
            stageA(nxt, k0 + 32);
            stageB(nxt, k0 + 32);
        }

        // ---- compute on current buffer ----
        const __bf16* A0 = &As[cur][(wave * 32 + lm) * 32];
        const __bf16* A1 = A0 + 16 * 32;
        v16bf a0 = *(const v16bf*)(A0 + half * 16);
        v16bf a1 = *(const v16bf*)(A1 + half * 16);
        #pragma unroll
        for (int j = 0; j < 4; ++j) {
            v16bf bb = *(const v16bf*)(&Bs[cur][(j * 16 + lm) * 32 + half * 16]);
            acc[0][j] = __builtin_amdgcn_wmma_f32_16x16x32_bf16(
                false, a0, false, bb, (short)0, acc[0][j], false, false);
            acc[1][j] = __builtin_amdgcn_wmma_f32_16x16x32_bf16(
                false, a1, false, bb, (short)0, acc[1][j], false, false);
        }

        if (wave == 0) __builtin_amdgcn_s_wait_tensorcnt(0);
        __syncthreads();
        cur = nxt;
    }

    // D layout: lane holds col n=lm, rows half*8 + r
    #pragma unroll
    for (int s = 0; s < 2; ++s) {
        const int lrow0 = wave * 32 + s * 16 + half * 8;
        #pragma unroll
        for (int j = 0; j < 4; ++j) {
            int gn = nbase + j * 16 + lm;
            if (gn < N) {
                #pragma unroll
                for (int r = 0; r < 8; ++r) {
                    int gm = mbase + lrow0 + r;
                    size_t oi = (size_t)b * M * N + (size_t)gm * N + gn;
                    float o = acc[s][j][r] + BiasLds[lrow0 + r];
                    if (resid) o += resid[oi];
                    O[oi] = o;
                }
            }
        }
    }
}

// =====================================================================
// Window mean-pool of q and k channels of qkv -> qr/kr as [b][r][c]
// =====================================================================
__global__ __launch_bounds__(256)
void pool_qk_kernel(const float* __restrict__ qkv,
                    float* __restrict__ qr, float* __restrict__ kr)
{
    int i = blockIdx.x * blockDim.x + threadIdx.x;
    if (i >= BATCH * RWIN * 1024) return;
    int c = i & 1023;
    int r = (i >> 10) % RWIN;
    int b = i / (RWIN * 1024);
    int y0 = (r / 7) * 4, x0 = (r % 7) * 4;
    const float* qb = qkv + (size_t)(b * 3072 + c) * NPIX;
    const float* kb = qkv + (size_t)(b * 3072 + 1024 + c) * NPIX;
    float sq = 0.f, sk = 0.f;
    #pragma unroll
    for (int p = 0; p < 4; ++p) {
        int off = (y0 + p) * HW + x0;
        float4 q4 = *(const float4*)(qb + off);
        float4 k4 = *(const float4*)(kb + off);
        sq += q4.x + q4.y + q4.z + q4.w;
        sk += k4.x + k4.y + k4.z + k4.w;
    }
    qr[(size_t)(b * RWIN + r) * 1024 + c] = sq * 0.0625f;
    kr[(size_t)(b * RWIN + r) * 1024 + c] = sk * 0.0625f;
}

// =====================================================================
// Routing: top-4 of a_r[b,i,:] ; one wave per (b,i)
// =====================================================================
__global__ __launch_bounds__(32)
void route_topk_kernel(const float* __restrict__ qr,
                       const float* __restrict__ kr,
                       int* __restrict__ idx)
{
    int u = blockIdx.x;
    int b = u / RWIN, r = u % RWIN;
    int lane = threadIdx.x;
    const float* q = qr + (size_t)(b * RWIN + r) * 1024;

    float v0 = -3.4e38f, v1 = -3.4e38f, v2 = -3.4e38f, v3 = -3.4e38f;
    int   i0 = 0, i1 = 0, i2 = 0, i3 = 0;

    for (int j = 0; j < RWIN; ++j) {
        const float* k = kr + (size_t)(b * RWIN + j) * 1024;
        float p = 0.f;
        for (int c = lane * 4; c < 1024; c += 128) {
            float4 qa = *(const float4*)(q + c);
            float4 ka = *(const float4*)(k + c);
            p += qa.x * ka.x + qa.y * ka.y + qa.z * ka.z + qa.w * ka.w;
        }
        #pragma unroll
        for (int off = 16; off > 0; off >>= 1) p += __shfl_xor(p, off, 32);
        if      (p > v0) { v3=v2;i3=i2; v2=v1;i2=i1; v1=v0;i1=i0; v0=p;i0=j; }
        else if (p > v1) { v3=v2;i3=i2; v2=v1;i2=i1; v1=p;i1=j; }
        else if (p > v2) { v3=v2;i3=i2; v2=p;i2=j; }
        else if (p > v3) { v3=p;i3=j; }
    }
    if (lane == 0) {
        int* o = idx + (size_t)u * 4;
        o[0] = i0; o[1] = i1; o[2] = i2; o[3] = i3;
    }
}

// =====================================================================
// BRA attention: one wave per (b, head, window r).
//   scores(16x64) = (q*scale)(16x128) @ kg^T     [16 wmma]
//   in-register softmax (shfl_xor across 16-lane halves)
//   out(16x128)   = attn(16x64) @ vg(64x128)     [16 wmma]
// =====================================================================
__global__ __launch_bounds__(32)
void bra_attn_kernel(const float* __restrict__ qkv,
                     const int* __restrict__ idx,
                     float* __restrict__ out)
{
    __shared__ __bf16 Qs[16 * 128];   // [s][d]
    __shared__ __bf16 Ks[64 * 128];   // [t][d]
    __shared__ __bf16 Vs[128 * 64];   // [d][t]
    __shared__ __bf16 Sa[16 * 64];    // attn probs, row-major

    int u = blockIdx.x;
    int r = u % RWIN;
    int t = u / RWIN;
    int head = t % NH;
    int b    = t / NH;

    int lane = threadIdx.x;
    int half = lane >> 4, lm = lane & 15;

    int wy0 = (r / 7) * 4, wx0 = (r % 7) * 4;
    const size_t base = (size_t)b * 3072 * NPIX;
    const int* widx = idx + (size_t)(b * RWIN + r) * 4;

    // ---- stage q: unit = (d, window-row); 4 contiguous pixels -> b128 ----
    for (int i = lane; i < 512; i += 32) {
        int d = i >> 2, srow = i & 3;
        const float* p = qkv + base + (size_t)(head * HD + d) * NPIX + (wy0 + srow) * HW + wx0;
        float4 v = *(const float4*)p;
        int s0 = srow * 4;
        Qs[(s0 + 0) * 128 + d] = (__bf16)(v.x * 0.03125f);
        Qs[(s0 + 1) * 128 + d] = (__bf16)(v.y * 0.03125f);
        Qs[(s0 + 2) * 128 + d] = (__bf16)(v.z * 0.03125f);
        Qs[(s0 + 3) * 128 + d] = (__bf16)(v.w * 0.03125f);
    }
    // ---- stage gathered k: unit = (d, which, srow) ----
    for (int i = lane; i < 2048; i += 32) {
        int d = i >> 4, tr = i & 15;
        int which = tr >> 2, srow = tr & 3;
        int wj = widx[which];
        const float* p = qkv + base + (size_t)(1024 + head * HD + d) * NPIX
                       + ((wj / 7) * 4 + srow) * HW + (wj % 7) * 4;
        float4 v = *(const float4*)p;
        int t0 = which * 16 + srow * 4;
        Ks[(t0 + 0) * 128 + d] = (__bf16)v.x;
        Ks[(t0 + 1) * 128 + d] = (__bf16)v.y;
        Ks[(t0 + 2) * 128 + d] = (__bf16)v.z;
        Ks[(t0 + 3) * 128 + d] = (__bf16)v.w;
    }
    // ---- stage gathered v transposed [d][t]: contiguous b64 LDS stores ----
    for (int i = lane; i < 2048; i += 32) {
        int d = i >> 4, tr = i & 15;
        int which = tr >> 2, srow = tr & 3;
        int wj = widx[which];
        const float* p = qkv + base + (size_t)(2048 + head * HD + d) * NPIX
                       + ((wj / 7) * 4 + srow) * HW + (wj % 7) * 4;
        float4 v = *(const float4*)p;
        v4bf o; o[0] = (__bf16)v.x; o[1] = (__bf16)v.y; o[2] = (__bf16)v.z; o[3] = (__bf16)v.w;
        *(v4bf*)(&Vs[d * 64 + which * 16 + srow * 4]) = o;
    }
    __syncthreads();

    // ---- scores = q @ k^T : M=16, N=64, K=128 ----
    v8f acc[4] = {};
    #pragma unroll
    for (int kk = 0; kk < 4; ++kk) {
        const __bf16* ap = &Qs[lm * 128 + kk * 32];
        v16bf a = *(const v16bf*)(ap + half * 16);
        #pragma unroll
        for (int j = 0; j < 4; ++j) {
            v16bf bb = *(const v16bf*)(&Ks[(j * 16 + lm) * 128 + kk * 32 + half * 16]);
            acc[j] = __builtin_amdgcn_wmma_f32_16x16x32_bf16(
                false, a, false, bb, (short)0, acc[j], false, false);
        }
    }

    // ---- softmax over 64 columns of each row ----
    #pragma unroll
    for (int rr = 0; rr < 8; ++rr) {
        float mx = fmaxf(fmaxf(acc[0][rr], acc[1][rr]),
                         fmaxf(acc[2][rr], acc[3][rr]));
        #pragma unroll
        for (int off = 8; off > 0; off >>= 1)
            mx = fmaxf(mx, __shfl_xor(mx, off, 32));
        float e0 = __expf(acc[0][rr] - mx);
        float e1 = __expf(acc[1][rr] - mx);
        float e2 = __expf(acc[2][rr] - mx);
        float e3 = __expf(acc[3][rr] - mx);
        float sum = e0 + e1 + e2 + e3;
        #pragma unroll
        for (int off = 8; off > 0; off >>= 1)
            sum += __shfl_xor(sum, off, 32);
        float inv = 1.0f / sum;
        int row = half * 8 + rr;
        Sa[row * 64 +  0 + lm] = (__bf16)(e0 * inv);
        Sa[row * 64 + 16 + lm] = (__bf16)(e1 * inv);
        Sa[row * 64 + 32 + lm] = (__bf16)(e2 * inv);
        Sa[row * 64 + 48 + lm] = (__bf16)(e3 * inv);
    }
    __syncthreads();

    // ---- out = attn @ vg : M=16, N=128, K=64 ----
    v8f acc2[8] = {};
    #pragma unroll
    for (int kk = 0; kk < 2; ++kk) {
        const __bf16* ap = &Sa[lm * 64 + kk * 32];
        v16bf a = *(const v16bf*)(ap + half * 16);
        #pragma unroll
        for (int j = 0; j < 8; ++j) {
            v16bf bb = *(const v16bf*)(&Vs[(j * 16 + lm) * 64 + kk * 32 + half * 16]);
            acc2[j] = __builtin_amdgcn_wmma_f32_16x16x32_bf16(
                false, a, false, bb, (short)0, acc2[j], false, false);
        }
    }

    // ---- store back to grid layout ----
    float* ob = out + (size_t)b * 1024 * NPIX;
    #pragma unroll
    for (int j = 0; j < 8; ++j) {
        int d = head * HD + j * 16 + lm;
        #pragma unroll
        for (int rr = 0; rr < 8; ++rr) {
            int s = half * 8 + rr;
            int y = wy0 + (s >> 2), x = wx0 + (s & 3);
            ob[(size_t)d * NPIX + y * HW + x] = acc2[j][rr];
        }
    }
}

// =====================================================================
// LEPE: depthwise 5x5 conv (pad 2) over v + bias, added to attention out
// =====================================================================
__global__ __launch_bounds__(256)
void lepe_add_kernel(const float* __restrict__ qkv,
                     const float* __restrict__ attn,
                     const float* __restrict__ wl,
                     const float* __restrict__ bl,
                     float* __restrict__ o1)
{
    int i = blockIdx.x * blockDim.x + threadIdx.x;
    if (i >= BATCH * 1024 * NPIX) return;
    int x = i % HW;
    int y = (i / HW) % HW;
    int c = (i / NPIX) & 1023;
    int b = i / (NPIX * 1024);
    const float* v = qkv + (size_t)(b * 3072 + 2048 + c) * NPIX;
    const float* w = wl + c * 25;
    float s = bl[c];
    #pragma unroll
    for (int dy = 0; dy < 5; ++dy) {
        int yy = y + dy - 2;
        if (yy < 0 || yy >= HW) continue;
        #pragma unroll
        for (int dx = 0; dx < 5; ++dx) {
            int xx = x + dx - 2;
            if (xx < 0 || xx >= HW) continue;
            s += v[yy * HW + xx] * w[dy * 5 + dx];
        }
    }
    o1[i] = attn[i] + s;
}

// =====================================================================
extern "C" void kernel_launch(void* const* d_in, const int* in_sizes, int n_in,
                              void* d_out, int out_size, void* d_ws, size_t ws_size,
                              hipStream_t stream)
{
    (void)in_sizes; (void)n_in; (void)out_size; (void)ws_size;

    const float* zf    = (const float*)d_in[0];
    const float* w2    = (const float*)d_in[1];
    const float* b2    = (const float*)d_in[2];
    const float* wqkv  = (const float*)d_in[3];
    const float* bqkv  = (const float*)d_in[4];
    const float* wlepe = (const float*)d_in[5];
    const float* blepe = (const float*)d_in[6];
    const float* wout  = (const float*)d_in[7];
    const float* bout  = (const float*)d_in[8];
    const float* w3    = (const float*)d_in[9];
    const float* b3    = (const float*)d_in[10];

    char* ws = (char*)d_ws;
    size_t off = 0;
    auto take = [&](size_t bytes) {
        char* p = ws + off;
        off = (off + bytes + 255) & ~(size_t)255;
        return p;
    };
    float*  xbuf    = (float*) take((size_t)BATCH * 1024 * NPIX * 4);  // x, later o1
    float*  qkvbuf  = (float*) take((size_t)BATCH * 3072 * NPIX * 4);
    float*  qrbuf   = (float*) take((size_t)BATCH * RWIN * 1024 * 4);
    float*  krbuf   = (float*) take((size_t)BATCH * RWIN * 1024 * 4);
    int*    idxbuf  = (int*)   take((size_t)BATCH * RWIN * 4 * 4);
    float*  attnbuf = (float*) take((size_t)BATCH * 1024 * NPIX * 4); // attn out, later y1
    __bf16* w2bf    = (__bf16*)take((size_t)1024 * 256 * 2);
    __bf16* wqkvbf  = (__bf16*)take((size_t)3072 * 1024 * 2);
    __bf16* woutbf  = (__bf16*)take((size_t)1024 * 1024 * 2);
    __bf16* w3bf    = (__bf16*)take((size_t)256 * 1024 * 2);

    dim3 blk(256);
    const int NB = (NPIX + 63) / 64;  // 13 N-tiles of 64

    // ---- pre-convert weights to bf16 (one-time, tiny) ----
    f32_to_bf16_kernel<<<(1024*256/8 + 255)/256, blk, 0, stream>>>(w2, w2bf, 1024*256);
    f32_to_bf16_kernel<<<(3072*1024/8 + 255)/256, blk, 0, stream>>>(wqkv, wqkvbf, 3072*1024);
    f32_to_bf16_kernel<<<(1024*1024/8 + 255)/256, blk, 0, stream>>>(wout, woutbf, 1024*1024);
    f32_to_bf16_kernel<<<(256*1024/8 + 255)/256, blk, 0, stream>>>(w3, w3bf, 256*1024);

    // x = conv1x1(zf, w2, b2)           [M=1024, K=256]
    wmma_gemm_kernel<<<dim3(NB, 1024 / 256, BATCH), blk, 0, stream>>>(
        w2bf, zf, b2, nullptr, xbuf, 1024, 256, NPIX);

    // qkv = conv1x1(x, wqkv, bqkv)      [M=3072, K=1024]
    wmma_gemm_kernel<<<dim3(NB, 3072 / 256, BATCH), blk, 0, stream>>>(
        wqkvbf, xbuf, bqkv, nullptr, qkvbuf, 3072, 1024, NPIX);

    // routing
    int npool = BATCH * RWIN * 1024;
    pool_qk_kernel<<<(npool + 255) / 256, blk, 0, stream>>>(qkvbuf, qrbuf, krbuf);
    route_topk_kernel<<<BATCH * RWIN, dim3(32), 0, stream>>>(qrbuf, krbuf, idxbuf);

    // gathered attention
    bra_attn_kernel<<<BATCH * NH * RWIN, dim3(32), 0, stream>>>(qkvbuf, idxbuf, attnbuf);

    // o1 = attn + lepe(v)   (o1 reuses xbuf)
    int nl = BATCH * 1024 * NPIX;
    lepe_add_kernel<<<(nl + 255) / 256, blk, 0, stream>>>(
        qkvbuf, attnbuf, wlepe, blepe, xbuf);

    // y1 = conv1x1(o1, w_out, b_out)    [M=1024, K=1024] (y1 reuses attnbuf)
    wmma_gemm_kernel<<<dim3(NB, 1024 / 256, BATCH), blk, 0, stream>>>(
        woutbf, xbuf, bout, nullptr, attnbuf, 1024, 1024, NPIX);

    // out = conv1x1(y1, w3, b3) + zf    [M=256, K=1024]
    wmma_gemm_kernel<<<dim3(NB, 256 / 256, BATCH), blk, 0, stream>>>(
        w3bf, attnbuf, b3, zf, (float*)d_out, 256, 1024, NPIX);
}
